// EncoderLayer_36739150250201
// MI455X (gfx1250) — matrix operand order repaired and checked by reading, not compiled, not run on hardware
//
#include <hip/hip_runtime.h>
#include <cstdint>
#include <cstddef>

// ---------------------------------------------------------------------------
// MI455X (gfx1250, CDNA5) transformer encoder layer.
// All GEMMs via v_wmma_f32_16x16x32_bf16 (wave32, 16x16 tiles, f32 accum).
// Flash attention: 64 keys/iteration (8 WMMAs per softmax pass), base-2
// online softmax with DPP16 VALU butterflies, LDS only for the P transpose.
// ---------------------------------------------------------------------------

typedef __attribute__((ext_vector_type(16))) __bf16 v16bf;
typedef __attribute__((ext_vector_type(8)))  __bf16 v8bf;
typedef __attribute__((ext_vector_type(8)))  float  v8f;

#define DEV static __device__ __forceinline__

// fp32 -> bf16 round-to-nearest-even (bit manipulation, no fptrunc dependence)
DEV __bf16 f2bf(float f) {
    unsigned u = __builtin_bit_cast(unsigned, f);
    u += 0x7fffu + ((u >> 16) & 1u);
    unsigned short h = (unsigned short)(u >> 16);
    return __builtin_bit_cast(__bf16, h);
}

// DPP16 lane permute within each 16-lane row (wave32: rows = lanes 0-15/16-31)
template<int CTRL>
DEV float dpp_f(float x) {
    return __builtin_bit_cast(float,
        __builtin_amdgcn_update_dpp(0, __builtin_bit_cast(int, x),
                                    CTRL, 0xf, 0xf, true));
}
// Butterfly reduction over the 16 lanes of each half-wave, pure VALU:
// quad_perm(1,0,3,2)=0xB1, quad_perm(2,3,0,1)=0x4E, row_half_mirror=0x141,
// row_mirror=0x140.  Result uniform across the 16 lanes.
DEV float rowmax16(float t) {
    t = fmaxf(t, dpp_f<0xB1>(t));
    t = fmaxf(t, dpp_f<0x4E>(t));
    t = fmaxf(t, dpp_f<0x141>(t));
    t = fmaxf(t, dpp_f<0x140>(t));
    return t;
}
DEV float rowsum16(float t) {
    t += dpp_f<0xB1>(t);
    t += dpp_f<0x4E>(t);
    t += dpp_f<0x141>(t);
    t += dpp_f<0x140>(t);
    return t;
}

// Load WMMA A-operand: per lane, 16 contiguous bf16 at p.
// (contiguous k order, matched by B: lane<16 holds K {0..15}; lane>=16 holds K {16..31};
//  caller supplies p already offset by aoff = (lane&16)?16:0.)
DEV v16bf ldA16(const __bf16* p) {
    v8bf lo = *(const v8bf*)(p);
    v8bf hi = *(const v8bf*)(p + 8);
    return __builtin_shufflevector(lo, hi, 0,1,2,3,4,5,6,7,8,9,10,11,12,13,14,15);
}

DEV v8f wmma_bf16(v16bf a, v16bf b, v8f c) {
    // (neg_a, A, neg_b, B, c_mod, C, reuse_a, reuse_b)
    return __builtin_amdgcn_wmma_f32_16x16x32_bf16(false, a, false, b,
                                                   (short)0, c, false, false);
}

// ---------------------------------------------------------------------------
// LayerNorm over D=256 per token, output bf16. One block (256 thr) per token.
// ---------------------------------------------------------------------------
__global__ __launch_bounds__(256)
void ln_to_bf16(const float* __restrict__ x, const float* __restrict__ g,
                const float* __restrict__ b, __bf16* __restrict__ out) {
    __shared__ float red[256];
    const int t = blockIdx.x, i = threadIdx.x;
    const float v = x[(size_t)t * 256 + i];
    red[i] = v; __syncthreads();
    #pragma unroll
    for (int s = 128; s > 0; s >>= 1) { if (i < s) red[i] += red[i + s]; __syncthreads(); }
    const float mu = red[0] * (1.0f / 256.0f);
    __syncthreads();
    const float d = v - mu;
    red[i] = d * d; __syncthreads();
    #pragma unroll
    for (int s = 128; s > 0; s >>= 1) { if (i < s) red[i] += red[i + s]; __syncthreads(); }
    const float var = red[0] * (1.0f / 256.0f);
    out[(size_t)t * 256 + i] = f2bf(d * rsqrtf(var + 1e-5f) * g[i] + b[i]);
}

// ---------------------------------------------------------------------------
// Repack row-major f32 weight [K, Nc] into WMMA B-operand layout (bf16):
// tile (kt, nt): for lane, element e:  n = nt*16 + (lane&15),
//   kk = kt*32 + ((lane&16)?16:0) + e   (contiguous k order, same as the A loads)
// Packed index = ((nt*KT + kt)*32 + lane)*16 + e  -> one 32B load per lane.
// ---------------------------------------------------------------------------
__global__ __launch_bounds__(256)
void pack_wmma_B(const float* __restrict__ W, __bf16* __restrict__ out,
                 int K, int Nc) {
    const int idx  = blockIdx.x * 256 + threadIdx.x;
    const int e    = idx & 15;
    const int lane = (idx >> 4) & 31;
    const int rest = idx >> 9;
    const int KT   = K >> 5;
    const int kt   = rest % KT;
    const int nt   = rest / KT;
    const int n    = (nt << 4) + (lane & 15);
    const int kk   = (kt << 5) + ((lane & 16) ? 16 : 0) + e;
    out[idx] = f2bf(W[(size_t)kk * Nc + n]);
}

// ---------------------------------------------------------------------------
// Generic WMMA GEMM: C[M,Nc] = A[M,K](bf16) @ Bpacked(bf16) + epilogue.
// Block = 128 thr (4 waves), WG tile 64x64, wave tile 32x32 (4 accumulators).
// ---------------------------------------------------------------------------
enum { EPI_QK = 0, EPI_VT = 1, EPI_RES = 2, EPI_GELU = 3 };

template<int EPI>
__global__ __launch_bounds__(128)
void gemm_wmma(const __bf16* __restrict__ A, const __bf16* __restrict__ Bp,
               const float* __restrict__ bias, const float* __restrict__ res,
               float* __restrict__ outF, __bf16* __restrict__ outB,
               int K, int Nc) {
    const int lane = threadIdx.x & 31;
    const int wv   = threadIdx.x >> 5;
    const int m0   = blockIdx.x * 64 + (wv >> 1) * 32;
    const int n0   = blockIdx.y * 64 + (wv & 1) * 32;
    const int KT   = K >> 5;
    const int l15  = lane & 15;
    const int aoff = (lane & 16) ? 16 : 0;

    v8f acc[2][2];
    acc[0][0] = {}; acc[0][1] = {}; acc[1][0] = {}; acc[1][1] = {};

    const __bf16* a0p = A + (size_t)(m0 + l15) * K + aoff;
    const __bf16* a1p = a0p + (size_t)16 * K;
    const __bf16* b0p = Bp + ((size_t)(n0 >> 4) * KT * 32 + lane) * 16;
    const __bf16* b1p = b0p + (size_t)KT * 512;

    for (int kt = 0; kt < KT; ++kt) {
        const int k0 = kt << 5;
        v16bf a0 = ldA16(a0p + k0);
        v16bf a1 = ldA16(a1p + k0);
        v16bf b0 = *(const v16bf*)(b0p + (size_t)kt * 512);
        v16bf b1 = *(const v16bf*)(b1p + (size_t)kt * 512);
        acc[0][0] = wmma_bf16(a0, b0, acc[0][0]);
        acc[0][1] = wmma_bf16(a0, b1, acc[0][1]);
        acc[1][0] = wmma_bf16(a1, b0, acc[1][0]);
        acc[1][1] = wmma_bf16(a1, b1, acc[1][1]);
    }

    // C/D layout: lane<16 -> rows 0..7 col lane; lane>=16 -> rows 8..15 col lane-16
    const int rb = (lane >> 4) * 8;
    #pragma unroll
    for (int mt = 0; mt < 2; ++mt)
    #pragma unroll
    for (int nt = 0; nt < 2; ++nt)
    #pragma unroll
    for (int i = 0; i < 8; ++i) {
        const int row = m0 + mt * 16 + rb + i;
        const int col = n0 + nt * 16 + l15;
        float v = acc[mt][nt][i] + bias[col];
        if constexpr (EPI == EPI_RES) {
            outF[(size_t)row * Nc + col] = v + res[(size_t)row * Nc + col];
        } else if constexpr (EPI == EPI_GELU) {
            const float ge = 0.5f * v * (1.0f + erff(v * 0.70710678118654752f));
            outB[(size_t)row * Nc + col] = f2bf(ge);
        } else {
            const int hh = col >> 5, d = col & 31;      // head, dim within head
            const int bb = row >> 11, nn = row & 2047;  // batch, seq pos (N=2048)
            if constexpr (EPI == EPI_QK)
                outB[(((size_t)(bb * 8 + hh) * 2048 + nn) << 5) + d] = f2bf(v);
            else  // EPI_VT: V transposed [B,H,32,N]
                outB[((size_t)(bb * 8 + hh) * 32 + d) * 2048 + nn] = f2bf(v);
        }
    }
}

// ---------------------------------------------------------------------------
// Flash attention, HD=32 (one WMMA K step). Block = 128 thr = 4 independent
// waves; each wave owns a 16-row Q tile of one (b,h) and streams keys SIXTY-
// FOUR at a time: 4 S WMMAs -> one base-2 online-softmax pass (DPP16 VALU
// butterflies, one O rescale) -> P transpose via LDS -> 4 P@V WMMAs.
// ---------------------------------------------------------------------------
__global__ __launch_bounds__(128)
void attn_flash(const __bf16* __restrict__ q, const __bf16* __restrict__ k,
                const __bf16* __restrict__ vt, const unsigned char* __restrict__ mask,
                __bf16* __restrict__ attnO) {
    __shared__ __bf16 plds[4][16 * 64];
    const int lane = threadIdx.x & 31;
    const int wv   = threadIdx.x >> 5;
    const int l15  = lane & 15;
    const int aoff = (lane & 16) ? 16 : 0;   // A-operand K-half offset
    const int boff = (lane & 16) ? 16 : 0;   // B-operand K-half offset
    const int bh = blockIdx.x;
    const int bb = bh >> 3, hh = bh & 7;
    const int q0 = (blockIdx.y * 4 + wv) << 4;

    const __bf16* qp = q  + (size_t)bh * 2048 * 32;
    const __bf16* kp = k  + (size_t)bh * 2048 * 32;
    const __bf16* vp = vt + (size_t)bh * 32 * 2048;
    __bf16* pl = plds[wv];

    const v16bf aQ = ldA16(qp + (size_t)(q0 + l15) * 32 + aoff);

    v8f o0 = {}, o1 = {};
    float mrow[8], lrow[8];
    #pragma unroll
    for (int i = 0; i < 8; ++i) { mrow[i] = -1e30f; lrow[i] = 0.0f; }
    // base-2 softmax: fold log2(e) into the 1/sqrt(HD) score scale
    const float sc2 = 0.17677669529663687f * 1.4426950408889634f;

    for (int j = 0; j < 2048; j += 64) {
        // K^T B-operands: column = key, K-dim = head dim (contiguous per key)
        const v16bf bK0 = *(const v16bf*)(kp + (size_t)(j      + l15) * 32 + boff);
        const v16bf bK1 = *(const v16bf*)(kp + (size_t)(j + 16 + l15) * 32 + boff);
        const v16bf bK2 = *(const v16bf*)(kp + (size_t)(j + 32 + l15) * 32 + boff);
        const v16bf bK3 = *(const v16bf*)(kp + (size_t)(j + 48 + l15) * 32 + boff);
        // prefetch next key/value tiles into GL2 (global_prefetch_b8)
        __builtin_prefetch(kp + (size_t)(j + 64 + l15) * 32, 0, 0);
        __builtin_prefetch(vp + (size_t)l15 * 2048 + j + 64, 0, 0);
        v8f z = {};
        v8f s0 = wmma_bf16(aQ, bK0, z);
        v8f s1 = wmma_bf16(aQ, bK1, z);
        v8f s2 = wmma_bf16(aQ, bK2, z);
        v8f s3 = wmma_bf16(aQ, bK3, z);

        const float mb0 = mask[bb * 2048 + j      + l15] ? -1e30f : 0.0f;
        const float mb1 = mask[bb * 2048 + j + 16 + l15] ? -1e30f : 0.0f;
        const float mb2 = mask[bb * 2048 + j + 32 + l15] ? -1e30f : 0.0f;
        const float mb3 = mask[bb * 2048 + j + 48 + l15] ? -1e30f : 0.0f;

        #pragma unroll
        for (int i = 0; i < 8; ++i) {
            float t0 = s0[i] * sc2 + mb0;
            float t1 = s1[i] * sc2 + mb1;
            float t2 = s2[i] * sc2 + mb2;
            float t3 = s3[i] * sc2 + mb3;
            // one butterfly per row per 64 keys (local 4-way max first)
            const float t = rowmax16(fmaxf(fmaxf(t0, t1), fmaxf(t2, t3)));
            const float mnew = fmaxf(mrow[i], t);
            const float corr = __builtin_amdgcn_exp2f(mrow[i] - mnew);
            const float p0 = __builtin_amdgcn_exp2f(t0 - mnew);
            const float p1 = __builtin_amdgcn_exp2f(t1 - mnew);
            const float p2 = __builtin_amdgcn_exp2f(t2 - mnew);
            const float p3 = __builtin_amdgcn_exp2f(t3 - mnew);
            const float ps = rowsum16((p0 + p1) + (p2 + p3));
            lrow[i] = lrow[i] * corr + ps;
            mrow[i] = mnew;
            o0[i] *= corr; o1[i] *= corr;
            // stash P (C layout -> row-major 16x64 in LDS)
            const int prow = (lane >> 4) * 8 + i;
            pl[prow * 64 + l15]      = f2bf(p0);
            pl[prow * 64 + l15 + 16] = f2bf(p1);
            pl[prow * 64 + l15 + 32] = f2bf(p2);
            pl[prow * 64 + l15 + 48] = f2bf(p3);
        }
        // reload P as two A-operands (transpose via LDS; same-wave DS in-order)
        const v16bf aP0 = ldA16(pl + l15 * 64 + aoff);
        const v16bf aP1 = ldA16(pl + l15 * 64 + 32 + aoff);
        // V^T B-operands: column = head dim, K-dim = key (V stored [B,H,32,N])
        const v16bf bV00 = *(const v16bf*)(vp + (size_t)l15        * 2048 + j      + boff);
        const v16bf bV01 = *(const v16bf*)(vp + (size_t)(16 + l15) * 2048 + j      + boff);
        const v16bf bV10 = *(const v16bf*)(vp + (size_t)l15        * 2048 + j + 32 + boff);
        const v16bf bV11 = *(const v16bf*)(vp + (size_t)(16 + l15) * 2048 + j + 32 + boff);
        o0 = wmma_bf16(aP0, bV00, o0);
        o1 = wmma_bf16(aP0, bV01, o1);
        o0 = wmma_bf16(aP1, bV10, o0);
        o1 = wmma_bf16(aP1, bV11, o1);
    }

    #pragma unroll
    for (int i = 0; i < 8; ++i) {
        const float inv = 1.0f / lrow[i];
        const int nn = q0 + (lane >> 4) * 8 + i;
        const size_t base = ((size_t)(bb * 2048 + nn)) * 256 + hh * 32;
        attnO[base + l15]      = f2bf(o0[i] * inv);
        attnO[base + 16 + l15] = f2bf(o1[i] * inv);
    }
}

// ---------------------------------------------------------------------------
// Host orchestration
// ---------------------------------------------------------------------------
extern "C" void kernel_launch(void* const* d_in, const int* in_sizes, int n_in,
                              void* d_out, int out_size, void* d_ws, size_t ws_size,
                              hipStream_t stream) {
    (void)in_sizes; (void)n_in; (void)out_size; (void)ws_size;
    const float*         tokens = (const float*)d_in[0];
    const unsigned char* kpm    = (const unsigned char*)d_in[1];  // bool mask
    const float* ln1_g = (const float*)d_in[2];
    const float* ln1_b = (const float*)d_in[3];
    const float* wq = (const float*)d_in[4];   const float* bq = (const float*)d_in[5];
    const float* wk = (const float*)d_in[6];   const float* bk = (const float*)d_in[7];
    const float* wv = (const float*)d_in[8];   const float* bv = (const float*)d_in[9];
    const float* wo = (const float*)d_in[10];  const float* bo = (const float*)d_in[11];
    const float* ln2_g = (const float*)d_in[12];
    const float* ln2_b = (const float*)d_in[13];
    const float* w1 = (const float*)d_in[14];  const float* b1 = (const float*)d_in[15];
    const float* w2 = (const float*)d_in[16];  const float* b2 = (const float*)d_in[17];
    float* out = (float*)d_out;

    // workspace carve-out (~50 MB total)
    char* w = (char*)d_ws;
    auto take = [&](size_t n) { char* r = w; w += (n + 255) & ~(size_t)255; return r; };
    __bf16* xn    = (__bf16*)take((size_t)8192 * 256 * 2);   // LN1(x), bf16
    __bf16* pq    = (__bf16*)take((size_t)256 * 256 * 2);    // packed weights
    __bf16* pk    = (__bf16*)take((size_t)256 * 256 * 2);
    __bf16* pv    = (__bf16*)take((size_t)256 * 256 * 2);
    __bf16* po    = (__bf16*)take((size_t)256 * 256 * 2);
    __bf16* p1w   = (__bf16*)take((size_t)256 * 1024 * 2);
    __bf16* p2w   = (__bf16*)take((size_t)1024 * 256 * 2);
    __bf16* qb    = (__bf16*)take((size_t)8192 * 256 * 2);   // Q  [B,H,N,32]
    __bf16* kb    = (__bf16*)take((size_t)8192 * 256 * 2);   // K  [B,H,N,32]
    __bf16* vtb   = (__bf16*)take((size_t)8192 * 256 * 2);   // V^T[B,H,32,N]
    __bf16* attnb = (__bf16*)take((size_t)8192 * 256 * 2);   // attn out, bf16
    float*  x1    = (float*) take((size_t)8192 * 256 * 4);   // x + attn proj, f32
    __bf16* xn2   = (__bf16*)take((size_t)8192 * 256 * 2);   // LN2(x1), bf16
    __bf16* hb    = (__bf16*)take((size_t)8192 * 1024 * 2);  // gelu hidden, bf16

    // 1) LN1 -> bf16
    ln_to_bf16<<<8192, 256, 0, stream>>>(tokens, ln1_g, ln1_b, xn);
    // 2) pack weights into WMMA B-operand layout
    pack_wmma_B<<<256,  256, 0, stream>>>(wq, pq,  256, 256);
    pack_wmma_B<<<256,  256, 0, stream>>>(wk, pk,  256, 256);
    pack_wmma_B<<<256,  256, 0, stream>>>(wv, pv,  256, 256);
    pack_wmma_B<<<256,  256, 0, stream>>>(wo, po,  256, 256);
    pack_wmma_B<<<1024, 256, 0, stream>>>(w1, p1w, 256, 1024);
    pack_wmma_B<<<1024, 256, 0, stream>>>(w2, p2w, 1024, 256);
    // 3) QKV projections (V written transposed)
    gemm_wmma<EPI_QK><<<dim3(128, 4),  128, 0, stream>>>(xn, pq, bq, nullptr, nullptr, qb,  256, 256);
    gemm_wmma<EPI_QK><<<dim3(128, 4),  128, 0, stream>>>(xn, pk, bk, nullptr, nullptr, kb,  256, 256);
    gemm_wmma<EPI_VT><<<dim3(128, 4),  128, 0, stream>>>(xn, pv, bv, nullptr, nullptr, vtb, 256, 256);
    // 4) flash attention: grid = (B*H, N/64), 4 waves x 16 q-rows per block
    attn_flash<<<dim3(32, 32), 128, 0, stream>>>(qb, kb, vtb, kpm, attnb);
    // 5) out projection + residual -> x1 (f32)
    gemm_wmma<EPI_RES><<<dim3(128, 4), 128, 0, stream>>>(attnb, po, bo, tokens, x1, nullptr, 256, 256);
    // 6) LN2 -> bf16
    ln_to_bf16<<<8192, 256, 0, stream>>>(x1, ln2_g, ln2_b, xn2);
    // 7) FFN up + exact-erf GELU -> bf16
    gemm_wmma<EPI_GELU><<<dim3(128, 16), 128, 0, stream>>>(xn2, p1w, b1, nullptr, nullptr, hb, 256, 1024);
    // 8) FFN down + residual -> final output (f32)
    gemm_wmma<EPI_RES><<<dim3(128, 4),  128, 0, stream>>>(hb, p2w, b2, x1, out, nullptr, 1024, 256);
}
